// LinearEncoder_12025908428993
// MI455X (gfx1250) — matrix operand order repaired and checked by reading, not compiled, not run on hardware
//
#include <hip/hip_runtime.h>
#include <hip/hip_bf16.h>
#include <stdint.h>

// ---- problem constants (from reference) ----
#define R_    8
#define C_    32
#define NPR_  128
#define RKEEP_ 6
#define HID_  1024
#define NLAT_ 160
#define B_    32
#define T_    1024
#define M_    (B_ * T_)      // 32768 tokens
#define K1_   NPR_           // 128
#define K2_   (R_ * C_)      // 256
#define K3_   HID_           // 1024

typedef __bf16 bf16_t;
typedef bf16_t v16bf __attribute__((ext_vector_type(16)));
typedef float  v8f   __attribute__((ext_vector_type(8)));
typedef int    v4i_  __attribute__((ext_vector_type(4)));

// -------- async global->LDS path (gfx1250), guarded so compile stays green --------
#if defined(__AMDGCN__) && __has_builtin(__builtin_amdgcn_global_load_async_to_lds_b128)
#define USE_ASYNC 1
#else
#define USE_ASYNC 0
#endif

#if USE_ASYNC
typedef __attribute__((address_space(1))) v4i_* g4p_t;  // global int4*
typedef __attribute__((address_space(3))) v4i_* l4p_t;  // LDS int4*
#endif

__device__ static inline void async_cp16(const bf16_t* g, bf16_t* l) {
#if USE_ASYNC
    __builtin_amdgcn_global_load_async_to_lds_b128((g4p_t)g, (l4p_t)l,
                                                   /*offset=*/0, /*cpol=*/0);
#else
    *(uint4*)l = *(const uint4*)g;
#endif
}

__device__ static inline void async_wait() {
#if USE_ASYNC
#if __has_builtin(__builtin_amdgcn_s_wait_asynccnt)
    __builtin_amdgcn_s_wait_asynccnt(0);
#else
    asm volatile("s_wait_asynccnt 0" ::: "memory");
#endif
#endif
}

// ---------------- WMMA helpers ----------------
__device__ static inline v8f wmma_bf16(v16bf a, v16bf b, v8f c) {
    // D = A(16x32 bf16) * B(32x16 bf16) + C(16x16 f32)
    return __builtin_amdgcn_wmma_f32_16x16x32_bf16(
        /*neg_a=*/false, a, /*neg_b=*/false, b,
        /*c_mod=*/(short)0, c, /*reuse_a=*/false, /*reuse_b=*/false);
}

// A fragment: 16x32 bf16, row-major A[m][k], row stride `stride` elements;
// `base` points at A[0][k0] of this 16x32 tile.
__device__ static inline v16bf load_frag_a(const bf16_t* base, int stride) {
    const int lane = threadIdx.x & 31;
    const int mrow = lane & 15;
    const int half = lane >> 4;
    const bf16_t* p = base + mrow * stride;
    v16bf f;
#pragma unroll
    for (int v = 0; v < 8; ++v) {
        const int k0 = half * 16 + 2 * v;
        f[2 * v]     = p[k0];
        f[2 * v + 1] = p[k0 + 1];
    }
    return f;
}

// B fragment: 32x16 bf16 stored TRANSPOSED as Bt[n][k] (k contiguous),
// row stride `stride`; `base` points at Bt[n0][k0] of this 16-col tile.
__device__ static inline v16bf load_frag_b(const bf16_t* base, int stride) {
    const int lane = threadIdx.x & 31;
    const int ncol = lane & 15;
    const int half = lane >> 4;
    const bf16_t* p = base + ncol * stride + half * 16;
    v16bf f;
#pragma unroll
    for (int v = 0; v < 8; ++v) {
        f[2 * v]     = p[2 * v];
        f[2 * v + 1] = p[2 * v + 1];
    }
    return f;
}

// ---------------- prep: f32 weights -> bf16 (identity layouts) ----------------
__global__ void prep_bf16(const float* __restrict__ W_area,
                          const float* __restrict__ U_W,
                          const float* __restrict__ V_W,
                          bf16_t* __restrict__ Wbf,   // [R][C][NPR] == Bt[c][n]
                          bf16_t* __restrict__ Ubf,   // [HID][K2]   == Bt[n][k]
                          bf16_t* __restrict__ Vbf) { // [NLAT][HID] == Bt[n][k]
    const int tid = blockIdx.x * blockDim.x + threadIdx.x;
    const int str = gridDim.x * blockDim.x;
    for (int i = tid; i < R_ * C_ * NPR_; i += str) Wbf[i] = (bf16_t)W_area[i];
    for (int i = tid; i < HID_ * K2_;    i += str) Ubf[i] = (bf16_t)U_W[i];
    for (int i = tid; i < NLAT_ * HID_;  i += str) Vbf[i] = (bf16_t)V_W[i];
}

// ---------------- k1: region embedding + mask -> h (bf16) ----------------
// One wave per 16-token tile; 8 waves / block. Per region: K=128 (4 WMMA
// k-steps), N=32 (2 WMMA column tiles).
__global__ void __launch_bounds__(256)
k1_region_emb(const float* __restrict__ spikes,
              const float* __restrict__ b_area,
              const int*   __restrict__ ids_shuffle,
              const bf16_t* __restrict__ Wbf,
              bf16_t* __restrict__ h_bf) {
    __shared__ bf16_t As[8][16 * K1_]; // per-wave 4KB staging, 32KB total

    const int wave = threadIdx.x >> 5;
    const int lane = threadIdx.x & 31;
    const int tile = blockIdx.x * 8 + wave;   // 0..2047
    const int m0   = tile * 16;               // token base (all same batch)
    const int b    = m0 >> 10;                // / T_

    unsigned keep = 0;
#pragma unroll
    for (int j = 0; j < RKEEP_; ++j) keep |= 1u << ids_shuffle[b * R_ + j];

    bf16_t* as = As[wave];
    const int half = lane >> 4;
    const int nc   = lane & 15;

    for (int r = 0; r < R_; ++r) {
        // pull next region's rows toward the WGP while this one computes
        if (r + 1 < R_ && lane < 16) {
            const float* nxt = spikes + (size_t)m0 * (R_ * NPR_) + (r + 1) * NPR_
                             + (size_t)lane * (R_ * NPR_);
            __builtin_prefetch(nxt, 0, 1);
        }
        if (!((keep >> r) & 1u)) {
            // masked region -> exact zeros (no bias)
            for (int i = lane; i < 16 * C_; i += 32) {
                const int mm = i >> 5, cc = i & 31;
                h_bf[(size_t)(m0 + mm) * K2_ + r * C_ + cc] = (bf16_t)0.0f;
            }
            continue;
        }
        // stage 16 x 128 f32 -> bf16 into LDS (float4 loads, coalesced)
        const float* src = spikes + (size_t)m0 * (R_ * NPR_) + r * NPR_;
        for (int i = lane; i < 16 * (NPR_ / 4); i += 32) {
            const int mm = i >> 5;      // row 0..15
            const int f4 = i & 31;      // float4 index 0..31
            const float4 v = ((const float4*)(src + (size_t)mm * (R_ * NPR_)))[f4];
            bf16_t* dst = as + mm * K1_ + f4 * 4;
            dst[0] = (bf16_t)v.x; dst[1] = (bf16_t)v.y;
            dst[2] = (bf16_t)v.z; dst[3] = (bf16_t)v.w;
        }
        // LDS same-wave ordering is in-order; wave-private region, no barrier
        v8f acc0 = {}; v8f acc1 = {};
        const bf16_t* wr = Wbf + r * (C_ * NPR_);
#pragma unroll
        for (int kk = 0; kk < K1_; kk += 32) {
            const v16bf a  = load_frag_a(as + kk, K1_);
            const v16bf b0 = load_frag_b(wr + kk, NPR_);             // cols 0..15
            const v16bf b1 = load_frag_b(wr + 16 * NPR_ + kk, NPR_); // cols 16..31
            acc0 = wmma_bf16(a, b0, acc0);
            acc1 = wmma_bf16(a, b1, acc1);
        }
        const float bias0 = b_area[r * C_ + nc];
        const float bias1 = b_area[r * C_ + 16 + nc];
#pragma unroll
        for (int i = 0; i < 8; ++i) {
            const int m = m0 + i + half * 8;
            h_bf[(size_t)m * K2_ + r * C_ + nc]      = (bf16_t)(acc0[i] + bias0);
            h_bf[(size_t)m * K2_ + r * C_ + 16 + nc] = (bf16_t)(acc1[i] + bias1);
        }
    }
}

// ---------------- k2: u = h @ U_W^T + U_b  (M=32768,K=256,N=1024) ----------------
// Block tile 64(M) x 128(N), 8 waves = 2(M) x 4(N), wave tile 32x32 (4 acc).
// Double-buffered LDS; next k-step staged via async global->LDS while the
// WMMA pipe works on the current buffer.
__global__ void __launch_bounds__(256)
k2_hidden(const bf16_t* __restrict__ h_bf,
          const bf16_t* __restrict__ Ubf,   // [n][k], k contiguous
          const float*  __restrict__ U_b,
          bf16_t* __restrict__ u_bf) {
    __shared__ bf16_t As[2][64 * 32];   // 2 x 4KB
    __shared__ bf16_t Bs[2][128 * 32];  // 2 x 8KB, stored as Bt[n][k]

    const int tid  = threadIdx.x;
    const int m0   = blockIdx.x * 64;
    const int n0   = blockIdx.y * 128;
    const int wave = tid >> 5, lane = tid & 31;
    const int wm   = wave & 1;       // 0..1
    const int wn   = wave >> 1;      // 0..3

    // per-thread staging addresses
    const int arow = tid >> 2, aseg = tid & 3;  // A: 64 rows, 16B/thread
    const int brow = tid >> 1, bseg = tid & 1;  // B: 128 rows, 32B/thread
    const bf16_t* ag = h_bf + (size_t)(m0 + arow) * K2_ + aseg * 8;
    const bf16_t* bg = Ubf + (size_t)(n0 + brow) * K2_ + bseg * 16;

    auto stage = [&](int kk, int buf) {
        async_cp16(ag + kk,     &As[buf][arow * 32 + aseg * 8]);
        async_cp16(bg + kk,     &Bs[buf][brow * 32 + bseg * 16]);
        async_cp16(bg + kk + 8, &Bs[buf][brow * 32 + bseg * 16 + 8]);
    };

    v8f acc[2][2] = {};

    stage(0, 0);
    async_wait();
    __syncthreads();

    for (int step = 0; step < K2_ / 32; ++step) {
        const int cur = step & 1;
        if (step + 1 < K2_ / 32) stage((step + 1) * 32, cur ^ 1);

        v16bf afrag[2], bfrag[2];
#pragma unroll
        for (int mi = 0; mi < 2; ++mi)
            afrag[mi] = load_frag_a(&As[cur][(wm * 32 + mi * 16) * 32], 32);
#pragma unroll
        for (int ni = 0; ni < 2; ++ni)
            bfrag[ni] = load_frag_b(&Bs[cur][(wn * 32 + ni * 16) * 32], 32);
#pragma unroll
        for (int mi = 0; mi < 2; ++mi)
#pragma unroll
            for (int ni = 0; ni < 2; ++ni)
                acc[mi][ni] = wmma_bf16(afrag[mi], bfrag[ni], acc[mi][ni]);

        async_wait();
        __syncthreads();
    }

    const int half = lane >> 4, nc = lane & 15;
#pragma unroll
    for (int mi = 0; mi < 2; ++mi) {
#pragma unroll
        for (int ni = 0; ni < 2; ++ni) {
            const int n = n0 + wn * 32 + ni * 16 + nc;
            const float bias = U_b[n];
#pragma unroll
            for (int i = 0; i < 8; ++i) {
                const int m = m0 + wm * 32 + mi * 16 + i + half * 8;
                u_bf[(size_t)m * HID_ + n] = (bf16_t)(acc[mi][ni][i] + bias);
            }
        }
    }
}

// ---------------- k3: out = u @ V_W^T + V_b  (M=32768,K=1024,N=160) ----------------
// Block tile 64(M) x 160(N), 8 waves = 4(M) x 2(N), wave tile 16x80 (5 acc).
// Same double-buffered async pipeline, 32 k-steps.
__global__ void __launch_bounds__(256)
k3_latent(const bf16_t* __restrict__ u_bf,
          const bf16_t* __restrict__ Vbf,   // [n][k], k contiguous
          const float*  __restrict__ V_b,
          float* __restrict__ out) {
    __shared__ bf16_t As[2][64 * 32];    // 2 x 4KB
    __shared__ bf16_t Bs[2][NLAT_ * 32]; // 2 x 10KB, Bt[n][k]

    const int tid  = threadIdx.x;
    const int m0   = blockIdx.x * 64;
    const int wave = tid >> 5, lane = tid & 31;
    const int wm   = wave >> 1;  // 0..3
    const int wn   = wave & 1;   // 0..1 -> n base 0 / 80

    const int arow = tid >> 2, aseg = tid & 3;
    const bf16_t* ag = u_bf + (size_t)(m0 + arow) * HID_ + aseg * 8;

    auto stage = [&](int kk, int buf) {
        async_cp16(ag + kk, &As[buf][arow * 32 + aseg * 8]);
        // B: 160 rows x 32 bf16 = 640 x 16B
        for (int idx = tid; idx < 640; idx += 256) {
            const int row = idx >> 2, seg = idx & 3;
            async_cp16(Vbf + (size_t)row * HID_ + kk + seg * 8,
                       &Bs[buf][row * 32 + seg * 8]);
        }
    };

    v8f acc[5] = {};

    stage(0, 0);
    async_wait();
    __syncthreads();

    for (int step = 0; step < K3_ / 32; ++step) {
        const int cur = step & 1;
        if (step + 1 < K3_ / 32) stage((step + 1) * 32, cur ^ 1);

        const v16bf a = load_frag_a(&As[cur][(wm * 16) * 32], 32);
#pragma unroll
        for (int ni = 0; ni < 5; ++ni) {
            const v16bf bb = load_frag_b(&Bs[cur][(wn * 80 + ni * 16) * 32], 32);
            acc[ni] = wmma_bf16(a, bb, acc[ni]);
        }

        async_wait();
        __syncthreads();
    }

    const int half = lane >> 4, nc = lane & 15;
#pragma unroll
    for (int ni = 0; ni < 5; ++ni) {
        const int n = wn * 80 + ni * 16 + nc;
        const float bias = V_b[n];
#pragma unroll
        for (int i = 0; i < 8; ++i) {
            const int m = m0 + wm * 16 + i + half * 8;
            out[(size_t)m * NLAT_ + n] = acc[ni][i] + bias;
        }
    }
}

// ---------------- launcher ----------------
extern "C" void kernel_launch(void* const* d_in, const int* in_sizes, int n_in,
                              void* d_out, int out_size, void* d_ws, size_t ws_size,
                              hipStream_t stream) {
    const float* spikes      = (const float*)d_in[0];
    const float* W_area      = (const float*)d_in[1];
    const float* b_area      = (const float*)d_in[2];
    const float* U_W         = (const float*)d_in[3];
    const float* U_b         = (const float*)d_in[4];
    const float* V_W         = (const float*)d_in[5];
    const float* V_b         = (const float*)d_in[6];
    const int*   ids_shuffle = (const int*)d_in[7];
    float* out = (float*)d_out;

    // workspace carve-up (256B aligned)
    uintptr_t p = (uintptr_t)d_ws;
    auto take = [&](size_t bytes) {
        uintptr_t cur = (p + 255) & ~(uintptr_t)255;
        p = cur + bytes;
        return (void*)cur;
    };
    bf16_t* Wbf  = (bf16_t*)take((size_t)R_ * C_ * NPR_ * 2);   //  64 KB
    bf16_t* Ubf  = (bf16_t*)take((size_t)HID_ * K2_ * 2);       // 512 KB
    bf16_t* Vbf  = (bf16_t*)take((size_t)NLAT_ * HID_ * 2);     // 320 KB
    bf16_t* h_bf = (bf16_t*)take((size_t)M_ * K2_ * 2);         //  16.8 MB
    bf16_t* u_bf = (bf16_t*)take((size_t)M_ * HID_ * 2);        //  67.1 MB
    (void)ws_size; (void)in_sizes; (void)n_in; (void)out_size;

    prep_bf16<<<256, 256, 0, stream>>>(W_area, U_W, V_W, Wbf, Ubf, Vbf);

    // k1: 2048 token tiles / 8 waves per block
    k1_region_emb<<<M_ / 16 / 8, 256, 0, stream>>>(spikes, b_area, ids_shuffle,
                                                   Wbf, h_bf);

    // k2: grid (M/64, N/128)
    k2_hidden<<<dim3(M_ / 64, HID_ / 128), 256, 0, stream>>>(h_bf, Ubf, U_b, u_bf);

    // k3: grid (M/64)
    k3_latent<<<M_ / 64, 256, 0, stream>>>(u_bf, Vbf, V_b, out);
}
